// TriXLinear_51934744543485
// MI455X (gfx1250) — matrix operand order repaired and checked by reading, not compiled, not run on hardware
//
#include <hip/hip_runtime.h>

// ---------------------------------------------------------------------------
// TriXLinear forward for MI455X (gfx1250, wave32, WMMA):
//   y = (x @ sign_ternary(W)^T) * scales * tile_gate
// B=4096, I=2048, O=8192, NUM_TILES=8 (gate tile width 1024)
//
// Pass 1: x f32 -> f16, W f32 -> ternary f16 (exact {-1,0,1}) into workspace.
// Pass 2: double-buffered LDS WMMA GEMM, 128x256 macro-tile, 64x64 wave tile,
//         v_wmma_f32_16x16x32_f16. Global->LDS staging uses the CDNA5 async
//         path (global_load_async_to_lds_b128 / ASYNCcnt), no VGPR staging.
// ---------------------------------------------------------------------------

typedef __attribute__((ext_vector_type(16))) _Float16 v16h;
typedef __attribute__((ext_vector_type(8)))  _Float16 v8h;
typedef __attribute__((ext_vector_type(8)))  float    v8f;
typedef int v4i __attribute__((vector_size(16)));   // matches builtin param type

#define BATCH   4096
#define IN_F    2048
#define OUT_F   8192

#define BLK_M 128
#define BLK_N 256
#define BLK_K 64
#define LDSS  72                 // 64 + 8 f16 pad: 144B rows, 16B-aligned b128 chunks
#define NKB   (IN_F / BLK_K)     // 32 K-slabs

#if defined(__has_builtin)
#if __has_builtin(__builtin_amdgcn_global_load_async_to_lds_b128)
#define HAVE_ASYNC_LDS 1
#endif
#endif
#ifndef HAVE_ASYNC_LDS
#define HAVE_ASYNC_LDS 0
#endif

#define AS1 __attribute__((address_space(1)))
#define AS3 __attribute__((address_space(3)))

// -------------------- pass 1: convert x to f16 -----------------------------
__global__ __launch_bounds__(256)
void cvt_x_f16(const float* __restrict__ x, _Float16* __restrict__ xh) {
    const size_t i = ((size_t)blockIdx.x * blockDim.x + threadIdx.x) * 8;
    const float4 a = *(const float4*)(x + i);
    const float4 b = *(const float4*)(x + i + 4);
    v8h o;
    o[0] = (_Float16)a.x; o[1] = (_Float16)a.y;
    o[2] = (_Float16)a.z; o[3] = (_Float16)a.w;
    o[4] = (_Float16)b.x; o[5] = (_Float16)b.y;
    o[6] = (_Float16)b.z; o[7] = (_Float16)b.w;
    *(v8h*)(xh + i) = o;
}

// -------------------- pass 1: ternarize W to f16 ---------------------------
__device__ __forceinline__ _Float16 tern(float v) {
    return (_Float16)(v > 0.5f ? 1.0f : (v < -0.5f ? -1.0f : 0.0f));
}

__global__ __launch_bounds__(256)
void tern_w_f16(const float* __restrict__ w, _Float16* __restrict__ wh) {
    const size_t i = ((size_t)blockIdx.x * blockDim.x + threadIdx.x) * 8;
    const float4 a = *(const float4*)(w + i);
    const float4 b = *(const float4*)(w + i + 4);
    v8h o;
    o[0] = tern(a.x); o[1] = tern(a.y); o[2] = tern(a.z); o[3] = tern(a.w);
    o[4] = tern(b.x); o[5] = tern(b.y); o[6] = tern(b.z); o[7] = tern(b.w);
    *(v8h*)(wh + i) = o;
}

// -------------------- async global -> LDS copy helper ----------------------
#if HAVE_ASYNC_LDS
__device__ __forceinline__ void async_cp_b128(const _Float16* g, _Float16* l) {
    __builtin_amdgcn_global_load_async_to_lds_b128(
        (AS1 v4i*)g, (AS3 v4i*)l, /*imm_offset=*/0, /*cpol=*/0);
}
__device__ __forceinline__ void wait_async0() {
#if __has_builtin(__builtin_amdgcn_s_wait_asynccnt)
    __builtin_amdgcn_s_wait_asynccnt(0);
#else
    asm volatile("s_wait_asynccnt 0x0" ::: "memory");
#endif
}
#endif

// -------------------- pass 2: WMMA GEMM + fused epilogue -------------------
__global__ __launch_bounds__(256)
void trix_gemm_wmma(const _Float16* __restrict__ xh,   // [BATCH, IN_F] f16
                    const _Float16* __restrict__ wh,   // [OUT_F, IN_F] ternary f16
                    const float*    __restrict__ scales,
                    const int*      __restrict__ gate, // [BATCH, 8]
                    float*          __restrict__ out)  // [BATCH, OUT_F]
{
    __shared__ _Float16 Asmem[2][BLK_M * LDSS];  // 2 x 18 KiB
    __shared__ _Float16 Bsmem[2][BLK_N * LDSS];  // 2 x 36 KiB

    const int tid   = threadIdx.x;
    const int lane  = tid & 31;
    const int wave  = tid >> 5;
    const int waveM = wave & 1;   // 2 waves along M: 64 rows each
    const int waveN = wave >> 1;  // 4 waves along N: 64 cols each
    const int half  = lane >> 4;  // lane half selects K sub-block per ISA layout
    const int l16   = lane & 15;

    const int bM = blockIdx.y * BLK_M;
    const int bN = blockIdx.x * BLK_N;

    // global<->LDS staging coords: thread covers (srow + i*32, sc8..sc8+7)
    const int srow = tid >> 3;        // 0..31
    const int sc8  = (tid & 7) * 8;   // 0,8,..,56

    v8f acc[4][4] = {}; // 4 M-tiles x 4 N-tiles of 16x16 f32

    // Fragment load helpers (ISA 16-bit A 16x32 / B 32x16 per-lane layouts).
    auto loadA = [&](const _Float16* Ab, int mt, int ko) -> v16h {
        const int row = waveM * 64 + mt * 16 + l16;
        const v8h lo = *(const v8h*)(&Ab[row * LDSS + ko + half * 8]);
        const v8h hi = *(const v8h*)(&Ab[row * LDSS + ko + 16 + half * 8]);
        return __builtin_shufflevector(lo, hi,
            0,1,2,3,4,5,6,7,8,9,10,11,12,13,14,15);
    };
    auto loadB = [&](const _Float16* Bb, int nt, int ko) -> v16h {
        const int row = waveN * 64 + nt * 16 + l16;
        const v8h lo = *(const v8h*)(&Bb[row * LDSS + ko + half * 8]);
        const v8h hi = *(const v8h*)(&Bb[row * LDSS + ko + 16 + half * 8]);
        return __builtin_shufflevector(lo, hi,
            0,1,2,3,4,5,6,7,8,9,10,11,12,13,14,15);
    };

#if HAVE_ASYNC_LDS
    // ---- async pipeline: cache -> LDS with no VGPR staging ----
    auto stage_async = [&](int buf, int kofs) {
        #pragma unroll
        for (int i = 0; i < 4; ++i)
            async_cp_b128(&xh[(size_t)(bM + srow + i * 32) * IN_F + kofs + sc8],
                          &Asmem[buf][(srow + i * 32) * LDSS + sc8]);
        #pragma unroll
        for (int i = 0; i < 8; ++i)
            async_cp_b128(&wh[(size_t)(bN + srow + i * 32) * IN_F + kofs + sc8],
                          &Bsmem[buf][(srow + i * 32) * LDSS + sc8]);
    };

    stage_async(0, 0);
    wait_async0();
    __syncthreads();
#else
    // ---- fallback: register-staged copy ----
    #pragma unroll
    for (int i = 0; i < 4; ++i) {
        const int row = srow + i * 32;
        *(v8h*)(&Asmem[0][row * LDSS + sc8]) =
            *(const v8h*)(&xh[(size_t)(bM + row) * IN_F + sc8]);
    }
    #pragma unroll
    for (int i = 0; i < 8; ++i) {
        const int row = srow + i * 32;
        *(v8h*)(&Bsmem[0][row * LDSS + sc8]) =
            *(const v8h*)(&wh[(size_t)(bN + row) * IN_F + sc8]);
    }
    __syncthreads();
#endif

    #pragma unroll 1
    for (int kb = 0; kb < NKB; ++kb) {
        const int cur   = kb & 1;
        const int nxt   = cur ^ 1;
        const int knext = (kb + 1) * BLK_K;
        const bool more = (kb + 1) < NKB;

#if HAVE_ASYNC_LDS
        // Issue next-slab async copies; they land in LDS during the WMMAs.
        if (more) stage_async(nxt, knext);
#else
        v8h ra[4], rb[8];
        if (more) {
            #pragma unroll
            for (int i = 0; i < 4; ++i)
                ra[i] = *(const v8h*)(&xh[(size_t)(bM + srow + i * 32) * IN_F + knext + sc8]);
            #pragma unroll
            for (int i = 0; i < 8; ++i)
                rb[i] = *(const v8h*)(&wh[(size_t)(bN + srow + i * 32) * IN_F + knext + sc8]);
        }
#endif

        // Compute on current buffer: 2 ko x 16 = 32 WMMAs per wave.
        const _Float16* Ab = Asmem[cur];
        const _Float16* Bb = Bsmem[cur];
        #pragma unroll
        for (int ko = 0; ko < BLK_K; ko += 32) {
            v16h afrag[4];
            #pragma unroll
            for (int mt = 0; mt < 4; ++mt) afrag[mt] = loadA(Ab, mt, ko);

            // Rotate B fragments: load bfrag[nt+1] before consuming bfrag[nt]
            // so no WMMA quad waits on loads issued in the same cycle.
            v16h bcur = loadB(Bb, 0, ko);
            #pragma unroll
            for (int nt = 0; nt < 4; ++nt) {
                v16h bnxt = bcur;
                if (nt < 3) bnxt = loadB(Bb, nt + 1, ko);
                #pragma unroll
                for (int mt = 0; mt < 4; ++mt) {
                    acc[mt][nt] = __builtin_amdgcn_wmma_f32_16x16x32_f16(
                        /*neg_a=*/false, afrag[mt],
                        /*neg_b=*/false, bcur,
                        /*c_mod=*/(short)0, acc[mt][nt],
                        /*reuse_a=*/false, /*reuse_b=*/false);
                }
                bcur = bnxt;
            }
        }

#if HAVE_ASYNC_LDS
        if (more) {
            wait_async0();       // this wave's async copies done
            __syncthreads();     // every wave's copies visible
        }
#else
        if (more) {
            #pragma unroll
            for (int i = 0; i < 4; ++i)
                *(v8h*)(&Asmem[nxt][(srow + i * 32) * LDSS + sc8]) = ra[i];
            #pragma unroll
            for (int i = 0; i < 8; ++i)
                *(v8h*)(&Bsmem[nxt][(srow + i * 32) * LDSS + sc8]) = rb[i];
            __syncthreads();
        }
#endif
    }

    // Epilogue: C/D layout — lane l16 = col, VGPR i = row (i + 8*half).
    // BLK_N=256 divides the 1024-wide gate tile, so t is workgroup-uniform.
    const int t = bN >> 10;
    float sc[4];
    #pragma unroll
    for (int nt = 0; nt < 4; ++nt)
        sc[nt] = scales[bN + waveN * 64 + nt * 16 + l16];

    #pragma unroll
    for (int mt = 0; mt < 4; ++mt) {
        #pragma unroll
        for (int i = 0; i < 8; ++i) {
            const int m = bM + waveM * 64 + mt * 16 + half * 8 + i;
            const float g = gate[m * 8 + t] ? 1.0f : 0.0f;
            #pragma unroll
            for (int nt = 0; nt < 4; ++nt) {
                const int n = bN + waveN * 64 + nt * 16 + l16;
                out[(size_t)m * OUT_F + n] = acc[mt][nt][i] * sc[nt] * g;
            }
        }
    }
}

// ---------------------------------------------------------------------------
extern "C" void kernel_launch(void* const* d_in, const int* in_sizes, int n_in,
                              void* d_out, int out_size, void* d_ws, size_t ws_size,
                              hipStream_t stream) {
    const float* x      = (const float*)d_in[0]; // [4096, 2048]
    const float* weight = (const float*)d_in[1]; // [8192, 2048]
    const float* scales = (const float*)d_in[2]; // [8192]
    const int*   gate   = (const int*)d_in[3];   // [4096, 8]
    float* out = (float*)d_out;                  // [4096, 8192]

    // Workspace layout: x_f16 (16 MiB) | w_ternary_f16 (32 MiB)
    char* ws = (char*)d_ws;
    _Float16* xh = (_Float16*)ws;
    _Float16* wh = (_Float16*)(ws + (size_t)BATCH * IN_F * sizeof(_Float16));

    // Pass 1: precision conversion (one touch of x and W each).
    {
        const int nx = BATCH * IN_F / 8;   // 8 elems per thread
        cvt_x_f16<<<nx / 256, 256, 0, stream>>>(x, xh);
        const int nw = OUT_F * IN_F / 8;
        tern_w_f16<<<nw / 256, 256, 0, stream>>>(weight, wh);
    }

    // Pass 2: tiled WMMA GEMM with fused scale + gate epilogue.
    {
        dim3 grid(OUT_F / BLK_N, BATCH / BLK_M); // (32, 32)
        trix_gemm_wmma<<<grid, 256, 0, stream>>>(xh, wh, scales, gate, out);
    }
}
